// Attention_27771258536213
// MI455X (gfx1250) — compile-verified, matrix operand order checked by reading, not run on hardware
//
#include <hip/hip_runtime.h>
#include <hip/hip_bf16.h>

// Multi-head attention forward for MI455X (gfx1250).
// bf16 WMMA pipeline, fully async (ASYNCcnt) double-buffered LDS staging.
// x:[4,2048,1024] f32; Wq/Wk/Wv/Wo:[1024,1024] f32; bo:[1024] f32 -> out f32.

#define EMBED    1024
#define HEADS    16
#define HEAD_DIM 64
#define NBATCH   4
#define SEQ      2048
#define ROWS     (NBATCH * SEQ)   // 8192

typedef __bf16 bf16_t;
typedef __bf16 v16bf __attribute__((ext_vector_type(16)));
typedef __bf16 v8bf  __attribute__((ext_vector_type(8)));
typedef float  v8f   __attribute__((ext_vector_type(8)));

// D = A(16x32 bf16) * B(32x16 bf16) + C(16x16 f32)
static __device__ __forceinline__ v8f wmma_bf16(v16bf a, v16bf b, v8f c) {
  return __builtin_amdgcn_wmma_f32_16x16x32_bf16(
      false, a, false, b, (short)0, c, false, false);
}

// Per-lane 16B async copy global -> LDS (GLOBAL_LOAD_ASYNC_TO_LDS_B128,
// ASYNCcnt). LDS offset = low 32 bits of the generic pointer.
static __device__ __forceinline__ void async_copy_b128(bf16_t* lds_ptr,
                                                       const bf16_t* gptr) {
  unsigned lds_off = (unsigned)(uintptr_t)lds_ptr;
  asm volatile("global_load_async_to_lds_b128 %0, %1, off"
               :: "v"(lds_off), "v"(gptr)
               : "memory");
}
static __device__ __forceinline__ void async_wait0() {
  asm volatile("s_wait_asynccnt 0" ::: "memory");
}
static __device__ __forceinline__ void async_wait4() {  // keep newest 4 in flight
  asm volatile("s_wait_asynccnt 4" ::: "memory");
}
static __device__ __forceinline__ void async_wait8() {  // keep newest 8 in flight
  asm volatile("s_wait_asynccnt 8" ::: "memory");
}

// 16x32 bf16 A-style fragment (or K-major B fragment) from LDS:
// lanes<16 take K {0..7,16..23}, lanes>=16 take {8..15,24..31}.
static __device__ __forceinline__ v16bf load_frag(const bf16_t* base, int lane) {
  const int sub = (lane >> 4) << 3;
  v8bf lo = *(const v8bf*)(base + sub);
  v8bf hi = *(const v8bf*)(base + sub + 16);
  return __builtin_shufflevector(lo, hi, 0, 1, 2, 3, 4, 5, 6, 7,
                                 8, 9, 10, 11, 12, 13, 14, 15);
}

// ---------------------------------------------------------------- converts
__global__ void cvt_f32_bf16(const float* __restrict__ src,
                             bf16_t* __restrict__ dst, int n) {
  int i = (blockIdx.x * blockDim.x + threadIdx.x) * 8;
  if (i < n) {
    float4 a = *(const float4*)(src + i);
    float4 b = *(const float4*)(src + i + 4);
    v8bf o;
    o[0] = (bf16_t)a.x; o[1] = (bf16_t)a.y; o[2] = (bf16_t)a.z; o[3] = (bf16_t)a.w;
    o[4] = (bf16_t)b.x; o[5] = (bf16_t)b.y; o[6] = (bf16_t)b.z; o[7] = (bf16_t)b.w;
    *(v8bf*)(dst + i) = o;
  }
}

// Transpose-convert: W f32 [E][E] -> Wt bf16 [E][E], Wt[c][r] = W[r][c].
// grid (E/64, E/64), block 256. Coalesced f32 reads, coalesced 16B bf16 writes.
__global__ __launch_bounds__(256) void cvt_transpose(const float* __restrict__ src,
                                                     bf16_t* __restrict__ dst) {
  __shared__ __align__(16) bf16_t tile[64 * 72];
  const int t = threadIdx.x;
  const int rb = blockIdx.y * 64, cb = blockIdx.x * 64;
  const int r0 = t >> 4, c0 = (t & 15) << 2;  // r0 in 0..15, c0 in 0..60
#pragma unroll
  for (int i = 0; i < 64; i += 16) {
    float4 a = *(const float4*)&src[(size_t)(rb + r0 + i) * EMBED + cb + c0];
    tile[(c0 + 0) * 72 + r0 + i] = (bf16_t)a.x;
    tile[(c0 + 1) * 72 + r0 + i] = (bf16_t)a.y;
    tile[(c0 + 2) * 72 + r0 + i] = (bf16_t)a.z;
    tile[(c0 + 3) * 72 + r0 + i] = (bf16_t)a.w;
  }
  __syncthreads();
  const int c1 = t >> 3, r1 = (t & 7) << 3;  // c1 in 0..31, r1 in 0..56
#pragma unroll
  for (int i = 0; i < 64; i += 32) {
    *(v8bf*)&dst[(size_t)(cb + c1 + i) * EMBED + rb + r1] =
        *(const v8bf*)&tile[(c1 + i) * 72 + r1];
  }
}

// ------------------------------------------------ shared GEMM tile staging
// 64x32 A tile + 64(cols)x32 Bt tile, 4 async b128 per thread (branchless).
static __device__ __forceinline__ void stage_gemm(bf16_t* sA, bf16_t* sBt,
                                                  const bf16_t* A,
                                                  const bf16_t* Bt, int rowBase,
                                                  int colBase, int k0, int t) {
  const int r = t >> 2, c = (t & 3) << 3;  // r in 0..31, c in {0,8,16,24}
  async_copy_b128(&sA[r * 40 + c],
                  &A[(size_t)(rowBase + r) * EMBED + k0 + c]);
  async_copy_b128(&sA[(r + 32) * 40 + c],
                  &A[(size_t)(rowBase + r + 32) * EMBED + k0 + c]);
  async_copy_b128(&sBt[r * 40 + c],
                  &Bt[(size_t)(colBase + r) * EMBED + k0 + c]);
  async_copy_b128(&sBt[(r + 32) * 40 + c],
                  &Bt[(size_t)(colBase + r + 32) * EMBED + k0 + c]);
}

// ------------------------------------------------------- QKV projection
// grid = (EMBED/64, ROWS/64, 3). 64x64 tile of x @ Wt[z]^T per block.
// z in {0,1}: write [N,H,S,D]; z==2 (V): write [N,H,D,S] (pre-transposed).
__global__ __launch_bounds__(128) void qkv_proj(const bf16_t* __restrict__ xb,
                                                const bf16_t* __restrict__ wtb,
                                                bf16_t* __restrict__ qkv) {
  __shared__ __align__(16) bf16_t sA[2][64 * 40];
  __shared__ __align__(16) bf16_t sBt[2][64 * 40];

  const int t = threadIdx.x;
  const int lane = t & 31;
  const int wave = t >> 5;
  const int rowBase = blockIdx.y * 64;
  const int colBase = blockIdx.x * 64;
  const bf16_t* Wt = wtb + (size_t)blockIdx.z * EMBED * EMBED;
  bf16_t* out = qkv + (size_t)blockIdx.z * (size_t)ROWS * EMBED;

  v8f acc[4] = {};

  stage_gemm(sA[0], sBt[0], xb, Wt, rowBase, colBase, 0, t);

  for (int k0 = 0; k0 < EMBED; k0 += 32) {
    const int buf = (k0 >> 5) & 1;
    __syncthreads();  // prior iteration's reads done before refilling buf^1
    if (k0 + 32 < EMBED) {
      stage_gemm(sA[buf ^ 1], sBt[buf ^ 1], xb, Wt, rowBase, colBase, k0 + 32, t);
      __builtin_prefetch(&xb[(size_t)(rowBase + (t >> 2)) * EMBED +
                             (k0 + 64 < EMBED ? k0 + 64 : 0)], 0, 1);
      async_wait4();  // drain current tile, keep next in flight
    } else {
      async_wait0();
    }
    __syncthreads();

    v16bf a = load_frag(&sA[buf][(16 * wave + (lane & 15)) * 40], lane);
#pragma unroll
    for (int f = 0; f < 4; ++f) {
      v16bf b = load_frag(&sBt[buf][(16 * f + (lane & 15)) * 40], lane);
      acc[f] = wmma_bf16(a, b, acc[f]);
    }
  }

  const int n0 = lane & 15;
  const int mhi = (lane >> 4) << 3;
#pragma unroll
  for (int f = 0; f < 4; ++f) {
#pragma unroll
    for (int j = 0; j < 8; ++j) {
      int grow = rowBase + 16 * wave + mhi + j;
      int gcol = colBase + 16 * f + n0;
      int nb = grow >> 11, s = grow & 2047;
      int h = gcol >> 6, d = gcol & 63;
      size_t head = (size_t)(nb * HEADS + h);
      if (blockIdx.z == 2)  // V stored [N,H,D,S] for async PV staging
        out[(head * HEAD_DIM + d) * SEQ + s] = (bf16_t)acc[f][j];
      else
        out[(head * SEQ + s) * HEAD_DIM + d] = (bf16_t)acc[f][j];
    }
  }
}

// ------------------------------------------------------ flash attention
// grid = (SEQ/64, NBATCH*HEADS). 4 waves; wave w owns q rows [16w,16w+16).
static __device__ __forceinline__ void stage_kv(bf16_t* sK, bf16_t* sVt,
                                                const bf16_t* K,
                                                const bf16_t* Vt, int kb, int t) {
  const int r = t >> 3, c = (t & 7) << 3;  // r in 0..15, c in {0..56}
#pragma unroll
  for (int i = 0; i < 64; i += 16)
    async_copy_b128(&sK[(r + i) * 72 + c],
                    &K[(size_t)(kb + r + i) * HEAD_DIM + c]);
#pragma unroll
  for (int i = 0; i < 64; i += 16)
    async_copy_b128(&sVt[(r + i) * 72 + c],
                    &Vt[(size_t)(r + i) * SEQ + kb + c]);
}

__global__ __launch_bounds__(128) void flash_attn(const bf16_t* __restrict__ qkv,
                                                  bf16_t* __restrict__ attnb) {
  __shared__ __align__(16) bf16_t sQ[64 * 72];        // [q][d]
  __shared__ __align__(16) bf16_t sK[2][64 * 72];     // [key][d]
  __shared__ __align__(16) bf16_t sVt[2][64 * 72];    // [d][key]
  __shared__ __align__(16) bf16_t sP[4][16 * 64];     // per-wave P scratch

  const int t = threadIdx.x;
  const int lane = t & 31;
  const int wave = t >> 5;
  const int nb = blockIdx.y >> 4;
  const int h = blockIdx.y & 15;
  const int qbase = blockIdx.x * 64;
  const size_t head = (size_t)(nb * HEADS + h);
  const bf16_t* Q = qkv + head * SEQ * HEAD_DIM;
  const bf16_t* K = qkv + (size_t)ROWS * EMBED + head * SEQ * HEAD_DIM;
  const bf16_t* Vt = qkv + 2 * (size_t)ROWS * EMBED + head * HEAD_DIM * SEQ;

  {  // Q tile: 4 async b128 per thread
    const int r = t >> 3, c = (t & 7) << 3;
#pragma unroll
    for (int i = 0; i < 64; i += 16)
      async_copy_b128(&sQ[(r + i) * 72 + c],
                      &Q[(size_t)(qbase + r + i) * HEAD_DIM + c]);
  }
  stage_kv(sK[0], sVt[0], K, Vt, 0, t);

  v8f o[4] = {};
  float mrow[8], lrow[8];
#pragma unroll
  for (int j = 0; j < 8; ++j) { mrow[j] = -1e30f; lrow[j] = 0.f; }

  const float scale = 0.125f;  // 1/sqrt(64)
  const int mhi = (lane >> 4) << 3;

  for (int kb = 0; kb < SEQ; kb += 64) {
    const int buf = (kb >> 6) & 1;
    __syncthreads();
    if (kb + 64 < SEQ) {
      stage_kv(sK[buf ^ 1], sVt[buf ^ 1], K, Vt, kb + 64, t);
      async_wait8();  // drain sQ + current K/V tile, keep next in flight
    } else {
      async_wait0();
    }
    __syncthreads();

    // scores: S = Q K^T, 16x64 per wave
    v8f sc[4] = {};
#pragma unroll
    for (int kk = 0; kk < 64; kk += 32) {
      v16bf a = load_frag(&sQ[(16 * wave + (lane & 15)) * 72 + kk], lane);
#pragma unroll
      for (int f = 0; f < 4; ++f) {
        v16bf b = load_frag(&sK[buf][(16 * f + (lane & 15)) * 72 + kk], lane);
        sc[f] = wmma_bf16(a, b, sc[f]);
      }
    }

    // online softmax; row (mhi+j) lives across a 16-lane half -> xor 1,2,4,8
#pragma unroll
    for (int j = 0; j < 8; ++j) {
      float mx = -1e30f;
#pragma unroll
      for (int f = 0; f < 4; ++f) mx = fmaxf(mx, sc[f][j]);
      mx = fmaxf(mx, __shfl_xor(mx, 1, 32));
      mx = fmaxf(mx, __shfl_xor(mx, 2, 32));
      mx = fmaxf(mx, __shfl_xor(mx, 4, 32));
      mx = fmaxf(mx, __shfl_xor(mx, 8, 32));
      mx *= scale;
      float mnew = fmaxf(mrow[j], mx);
      float alpha = __expf(mrow[j] - mnew);
      mrow[j] = mnew;
      float rsum = 0.f;
#pragma unroll
      for (int f = 0; f < 4; ++f) {
        float p = __expf(sc[f][j] * scale - mnew);
        sc[f][j] = p;
        rsum += p;
      }
      rsum += __shfl_xor(rsum, 1, 32);
      rsum += __shfl_xor(rsum, 2, 32);
      rsum += __shfl_xor(rsum, 4, 32);
      rsum += __shfl_xor(rsum, 8, 32);
      lrow[j] = lrow[j] * alpha + rsum;
#pragma unroll
      for (int f = 0; f < 4; ++f) o[f][j] *= alpha;
    }

    // transpose P (C layout) -> A layout through per-wave LDS scratch
#pragma unroll
    for (int f = 0; f < 4; ++f)
#pragma unroll
      for (int j = 0; j < 8; ++j)
        sP[wave][(mhi + j) * 64 + 16 * f + (lane & 15)] = (bf16_t)sc[f][j];
    __syncthreads();

    // O += P V
#pragma unroll
    for (int kk = 0; kk < 64; kk += 32) {
      v16bf a = load_frag(&sP[wave][(lane & 15) * 64 + kk], lane);
#pragma unroll
      for (int f = 0; f < 4; ++f) {
        v16bf b = load_frag(&sVt[buf][(16 * f + (lane & 15)) * 72 + kk], lane);
        o[f] = wmma_bf16(a, b, o[f]);
      }
    }
  }

  // normalize and store attn output as bf16 in [N, S, E]
#pragma unroll
  for (int j = 0; j < 8; ++j) {
    float inv = 1.f / lrow[j];
#pragma unroll
    for (int f = 0; f < 4; ++f) {
      int s = qbase + 16 * wave + mhi + j;
      int col = h * HEAD_DIM + 16 * f + (lane & 15);
      attnb[((size_t)nb * SEQ + s) * EMBED + col] = (bf16_t)(o[f][j] * inv);
    }
  }
}

// ----------------------------------------------------- output projection
__global__ __launch_bounds__(128) void out_proj(const bf16_t* __restrict__ attnb,
                                                const bf16_t* __restrict__ wot,
                                                const float* __restrict__ bo,
                                                float* __restrict__ out) {
  __shared__ __align__(16) bf16_t sA[2][64 * 40];
  __shared__ __align__(16) bf16_t sBt[2][64 * 40];

  const int t = threadIdx.x;
  const int lane = t & 31;
  const int wave = t >> 5;
  const int rowBase = blockIdx.y * 64;
  const int colBase = blockIdx.x * 64;

  v8f acc[4] = {};
  stage_gemm(sA[0], sBt[0], attnb, wot, rowBase, colBase, 0, t);

  for (int k0 = 0; k0 < EMBED; k0 += 32) {
    const int buf = (k0 >> 5) & 1;
    __syncthreads();
    if (k0 + 32 < EMBED) {
      stage_gemm(sA[buf ^ 1], sBt[buf ^ 1], attnb, wot, rowBase, colBase,
                 k0 + 32, t);
      async_wait4();
    } else {
      async_wait0();
    }
    __syncthreads();

    v16bf a = load_frag(&sA[buf][(16 * wave + (lane & 15)) * 40], lane);
#pragma unroll
    for (int f = 0; f < 4; ++f) {
      v16bf b = load_frag(&sBt[buf][(16 * f + (lane & 15)) * 40], lane);
      acc[f] = wmma_bf16(a, b, acc[f]);
    }
  }

  const int n0 = lane & 15;
  const int mhi = (lane >> 4) << 3;
#pragma unroll
  for (int f = 0; f < 4; ++f) {
#pragma unroll
    for (int j = 0; j < 8; ++j) {
      int grow = rowBase + 16 * wave + mhi + j;
      int gcol = colBase + 16 * f + n0;
      out[(size_t)grow * EMBED + gcol] = acc[f][j] + bo[gcol];
    }
  }
}

// ---------------------------------------------------------------- launch
extern "C" void kernel_launch(void* const* d_in, const int* in_sizes, int n_in,
                              void* d_out, int out_size, void* d_ws,
                              size_t ws_size, hipStream_t stream) {
  const float* x  = (const float*)d_in[0];
  const float* Wq = (const float*)d_in[1];
  const float* Wk = (const float*)d_in[2];
  const float* Wv = (const float*)d_in[3];
  const float* Wo = (const float*)d_in[4];
  const float* bo = (const float*)d_in[5];
  float* out = (float*)d_out;

  // workspace (bytes):
  //   xb    @ 0         : 16,777,216   (x bf16)
  //   wtb   @ 16777216  :  8,388,608   (Wq^T,Wk^T,Wv^T,Wo^T bf16)
  //   qkvb  @ 25165824  : 50,331,648   (Q,K:[N,H,S,D]; V:[N,H,D,S] bf16)
  //   attnb @ 75497472  : 16,777,216   (attn out bf16 [N,S,E])
  char* ws = (char*)d_ws;
  bf16_t* xb = (bf16_t*)(ws);
  bf16_t* wtb = (bf16_t*)(ws + 16777216ull);
  bf16_t* wot = wtb + 3ull * EMBED * EMBED;
  bf16_t* qkvb = (bf16_t*)(ws + 25165824ull);
  bf16_t* attnb = (bf16_t*)(ws + 75497472ull);

  const int nX = ROWS * EMBED;
  cvt_f32_bf16<<<(nX / 8 + 255) / 256, 256, 0, stream>>>(x, xb, nX);
  dim3 tg(EMBED / 64, EMBED / 64);
  cvt_transpose<<<tg, 256, 0, stream>>>(Wq, wtb + 0ull * EMBED * EMBED);
  cvt_transpose<<<tg, 256, 0, stream>>>(Wk, wtb + 1ull * EMBED * EMBED);
  cvt_transpose<<<tg, 256, 0, stream>>>(Wv, wtb + 2ull * EMBED * EMBED);
  cvt_transpose<<<tg, 256, 0, stream>>>(Wo, wot);

  qkv_proj<<<dim3(EMBED / 64, ROWS / 64, 3), 128, 0, stream>>>(xb, wtb, qkvb);
  flash_attn<<<dim3(SEQ / 64, NBATCH * HEADS), 128, 0, stream>>>(qkvb, attnb);
  out_proj<<<dim3(EMBED / 64, ROWS / 64), 128, 0, stream>>>(attnb, wot, bo, out);
}
